// MoELayer_45612552683585
// MI455X (gfx1250) — compile-verified, matrix operand order checked by reading, not run on hardware
//
#include <hip/hip_runtime.h>
#include <hip/hip_bf16.h>
#include <math.h>

// MoE top-2: router -> grouped (expert-sorted) bf16 WMMA GEMM1+GELU -> WMMA GEMM2
// -> weighted atomic combine. Workspace use: ~64.2 MB (h intermediate in bf16).
// Each wave computes a 2x2 grid of 16x16 WMMA tiles (4 wmma per 4 frag loads).

#define TOKENS   4096
#define H_DIM    1024
#define F_DIM    4096
#define NE       8
#define ROWS_TOTAL (TOKENS * 2)     // 8192 (token, expert) assignments
#define TILE_M   64
#define TILE_N   128
#define KCHUNK   128
#define LDK      (KCHUNK + 8)       // 136 elems; 272B row stride keeps 16B align

typedef __bf16 bf16_t;
typedef __attribute__((ext_vector_type(16))) __bf16 v16bf;
typedef __attribute__((ext_vector_type(8)))  __bf16 v8bf;
typedef __attribute__((ext_vector_type(8)))  float  v8f;

#if __has_builtin(__builtin_amdgcn_global_load_async_to_lds_b128) && \
    __has_builtin(__builtin_amdgcn_s_wait_asynccnt)
#define HAVE_ASYNC_LDS 1
typedef int v4i_t __attribute__((vector_size(16)));
typedef __attribute__((address_space(1))) v4i_t* gptr_b128;
typedef __attribute__((address_space(3))) v4i_t* lptr_b128;
#else
#define HAVE_ASYNC_LDS 0
#endif

static __device__ __forceinline__ bf16_t f2bf(float f) { return (bf16_t)f; }

// Build a 16x32 bf16 fragment (A-layout; B uses same layout with tile stored
// transposed [N][K] in LDS). ISA 16-bit layout: lanes 0-15 hold K=base+0..7 and
// base+16..23; lanes 16-31 hold K=base+8..15 and base+24..31. Two ds_load_b128.
static __device__ __forceinline__ v16bf frag16(const bf16_t* tile, int lane, int k0) {
    const bf16_t* p = tile + (lane & 15) * LDK + k0 + ((lane >> 4) << 3);
    v8bf lo = *(const v8bf*)p;
    v8bf hi = *(const v8bf*)(p + 16);
    return __builtin_shufflevector(lo, hi, 0,1,2,3,4,5,6,7,8,9,10,11,12,13,14,15);
}

#define WMMA_BF16(A, B, C) \
    __builtin_amdgcn_wmma_f32_16x16x32_bf16(false, (A), false, (B), (short)0, (C), false, false)

// ---------------- Router: 1 wave per token ----------------
__global__ __launch_bounds__(256) void router_k(const float* __restrict__ hs,
                                                const float* __restrict__ rw,
                                                const float* __restrict__ rb,
                                                int* __restrict__ tokExp,
                                                float* __restrict__ tokW) {
    int wave = threadIdx.x >> 5, lane = threadIdx.x & 31;
    int token = blockIdx.x * 8 + wave;
    if (token >= TOKENS) return;
    const float* x = hs + (size_t)token * H_DIM;
    float acc[NE];
#pragma unroll
    for (int e = 0; e < NE; e++) acc[e] = 0.f;
    for (int h = lane; h < H_DIM; h += 32) {
        float xv = x[h];
        const float4 r0 = *(const float4*)(rw + h * NE);
        const float4 r1 = *(const float4*)(rw + h * NE + 4);
        acc[0] += xv * r0.x; acc[1] += xv * r0.y; acc[2] += xv * r0.z; acc[3] += xv * r0.w;
        acc[4] += xv * r1.x; acc[5] += xv * r1.y; acc[6] += xv * r1.z; acc[7] += xv * r1.w;
    }
#pragma unroll
    for (int e = 0; e < NE; e++)
#pragma unroll
        for (int m = 16; m >= 1; m >>= 1) acc[e] += __shfl_xor(acc[e], m, 32);
    if (lane == 0) {
        float l[NE], mx = -1e30f;
#pragma unroll
        for (int e = 0; e < NE; e++) { l[e] = acc[e] + rb[e]; mx = fmaxf(mx, l[e]); }
        float ex[NE];
#pragma unroll
        for (int e = 0; e < NE; e++) ex[e] = __expf(l[e] - mx);
        int i0 = 0; float p0 = ex[0];
#pragma unroll
        for (int e = 1; e < NE; e++) if (ex[e] > p0) { p0 = ex[e]; i0 = e; }
        int i1 = -1; float p1 = -1.f;
#pragma unroll
        for (int e = 0; e < NE; e++) if (e != i0 && ex[e] > p1) { p1 = ex[e]; i1 = e; }
        float s = p0 + p1;
        tokExp[token * 2 + 0] = i0; tokExp[token * 2 + 1] = i1;
        tokW[token * 2 + 0] = p0 / s; tokW[token * 2 + 1] = p1 / s;
    }
}

// ---------------- Grouped-GEMM bookkeeping ----------------
__global__ void zero_counts_k(int* cnt, int* cnt2) {
    if (threadIdx.x < NE) { cnt[threadIdx.x] = 0; cnt2[threadIdx.x] = 0; }
}
__global__ void count_k(const int* __restrict__ tokExp, int* __restrict__ cnt) {
    int t = blockIdx.x * 256 + threadIdx.x;
    if (t < ROWS_TOTAL) atomicAdd(&cnt[tokExp[t]], 1);
}
__global__ void prefix_k(const int* __restrict__ cnt, int* __restrict__ eoff) {
    if (threadIdx.x == 0) { int r = 0; for (int e = 0; e < NE; e++) { eoff[e] = r; r += cnt[e]; } }
}
__global__ void scatter_k(const int* __restrict__ tokExp, const float* __restrict__ tokW,
                          const int* __restrict__ eoff, int* __restrict__ cnt2,
                          int* __restrict__ rowTok, float* __restrict__ rowW) {
    int t = blockIdx.x * 256 + threadIdx.x;
    if (t >= ROWS_TOTAL) return;
    int e = tokExp[t];
    int slot = atomicAdd(&cnt2[e], 1);
    int row = eoff[e] + slot;
    rowTok[row] = t >> 1;
    rowW[row] = tokW[t];
}
__global__ void zero_out_k(float* out) {
    size_t i = (size_t)blockIdx.x * 256 + threadIdx.x;
    if (i < (size_t)TOKENS * H_DIM) out[i] = 0.f;
}

// ---------------- FFN1: h = gelu(x @ w1[e] + b1[e]) -> bf16 ----------------
__global__ __launch_bounds__(256) void ffn1_k(const float* __restrict__ hs,
                                              const float* __restrict__ w1,
                                              const float* __restrict__ b1,
                                              const int* __restrict__ eoff,
                                              const int* __restrict__ cnt,
                                              const int* __restrict__ rowTok,
                                              bf16_t* __restrict__ hbuf) {
    int e = blockIdx.x >> 6, mt = blockIdx.x & 63;
    int rowBase = eoff[e] + mt * TILE_M;
    int rowEnd  = eoff[e] + cnt[e];
    if (rowBase >= rowEnd) return;
    int n0 = blockIdx.y * TILE_N;
    const float* w1e = w1 + (size_t)e * H_DIM * F_DIM;

    __shared__ __attribute__((aligned(16))) bf16_t As[TILE_M][LDK];
    __shared__ __attribute__((aligned(16))) bf16_t Bs[TILE_N][LDK];   // transposed [N][K]

    int tid = threadIdx.x, wave = tid >> 5, lane = tid & 31;
    int mi0 = (wave & 1) << 1;        // {0,2}   -> rows mi0, mi0+1
    int ni0 = (wave >> 1) << 1;       // {0..6}  -> cols ni0, ni0+1
    v8f acc00 = {}, acc01 = {}, acc10 = {}, acc11 = {};

    for (int kc = 0; kc < H_DIM; kc += KCHUNK) {
        if (kc + KCHUNK < H_DIM)
            __builtin_prefetch(w1e + (size_t)(kc + KCHUNK) * F_DIM + n0, 0, 1);
        // Stage A (tokens gathered by row list), fp32 -> bf16. float4 per task.
        for (int idx = tid; idx < TILE_M * (KCHUNK / 4); idx += 256) {
            int r = idx >> 5, g = idx & 31;
            int gr = rowBase + r;
            int tok = rowTok[gr < rowEnd ? gr : (rowEnd - 1)];
            const float4 v = *(const float4*)(hs + (size_t)tok * H_DIM + kc + g * 4);
            bf16_t* d = &As[r][g * 4];
            d[0] = f2bf(v.x); d[1] = f2bf(v.y); d[2] = f2bf(v.z); d[3] = f2bf(v.w);
        }
        // Stage B transposed: w1e[k][n] -> Bs[n][k], fp32 -> bf16.
        for (int idx = tid; idx < KCHUNK * (TILE_N / 4); idx += 256) {
            int k = idx >> 5, g = idx & 31;
            const float4 v = *(const float4*)(w1e + (size_t)(kc + k) * F_DIM + n0 + g * 4);
            Bs[g * 4 + 0][k] = f2bf(v.x); Bs[g * 4 + 1][k] = f2bf(v.y);
            Bs[g * 4 + 2][k] = f2bf(v.z); Bs[g * 4 + 3][k] = f2bf(v.w);
        }
        __syncthreads();
#pragma unroll
        for (int ks = 0; ks < KCHUNK; ks += 32) {
            v16bf a0 = frag16(&As[mi0 * 16][0], lane, ks);
            v16bf a1 = frag16(&As[(mi0 + 1) * 16][0], lane, ks);
            v16bf b0 = frag16(&Bs[ni0 * 16][0], lane, ks);
            v16bf b1f = frag16(&Bs[(ni0 + 1) * 16][0], lane, ks);
            acc00 = WMMA_BF16(a0, b0, acc00);
            acc01 = WMMA_BF16(a0, b1f, acc01);
            acc10 = WMMA_BF16(a1, b0, acc10);
            acc11 = WMMA_BF16(a1, b1f, acc11);
        }
        __syncthreads();
    }
    int mrow0 = mi0 * 16 + (lane >> 4) * 8;
    int nc0 = n0 + ni0 * 16 + (lane & 15);
    v8f accs[2][2] = {{acc00, acc01}, {acc10, acc11}};
#pragma unroll
    for (int i = 0; i < 2; i++)
#pragma unroll
        for (int j = 0; j < 2; j++) {
            int nc = nc0 + j * 16;
            float bias = b1[e * F_DIM + nc];
#pragma unroll
            for (int v = 0; v < 8; v++) {
                int gr = rowBase + mrow0 + i * 16 + v;
                if (gr < rowEnd) {
                    float x = accs[i][j][v] + bias;
                    hbuf[(size_t)gr * F_DIM + nc] =
                        f2bf(0.5f * x * (1.f + erff(x * 0.70710678f)));
                }
            }
        }
}

// ---------------- FFN2: out[tok] += w * (h @ w2[e] + b2[e]) ----------------
__global__ __launch_bounds__(256) void ffn2_k(const bf16_t* __restrict__ hbuf,
                                              const float* __restrict__ w2,
                                              const float* __restrict__ b2,
                                              const int* __restrict__ eoff,
                                              const int* __restrict__ cnt,
                                              const int* __restrict__ rowTok,
                                              const float* __restrict__ rowW,
                                              float* __restrict__ out) {
    int e = blockIdx.x >> 6, mt = blockIdx.x & 63;
    int rowBase = eoff[e] + mt * TILE_M;
    int rowEnd  = eoff[e] + cnt[e];
    if (rowBase >= rowEnd) return;
    int n0 = blockIdx.y * TILE_N;
    const float* w2e = w2 + (size_t)e * F_DIM * H_DIM;

    __shared__ __attribute__((aligned(16))) bf16_t As[TILE_M][LDK];
    __shared__ __attribute__((aligned(16))) bf16_t Bs[TILE_N][LDK];

    int tid = threadIdx.x, wave = tid >> 5, lane = tid & 31;
    int mi0 = (wave & 1) << 1;
    int ni0 = (wave >> 1) << 1;
    v8f acc00 = {}, acc01 = {}, acc10 = {}, acc11 = {};

    for (int kc = 0; kc < F_DIM; kc += KCHUNK) {
        if (kc + KCHUNK < F_DIM)
            __builtin_prefetch(w2e + (size_t)(kc + KCHUNK) * H_DIM + n0, 0, 1);
        // Stage A (already bf16, contiguous K): async global->LDS if available.
        for (int idx = tid; idx < TILE_M * (KCHUNK / 8); idx += 256) {
            int r = idx >> 4, g = idx & 15;
            int gr = rowBase + r;
            int src = (gr < rowEnd ? gr : rowEnd - 1);
            const bf16_t* gp = hbuf + (size_t)src * F_DIM + kc + g * 8;
            bf16_t* lp = &As[r][g * 8];
#if HAVE_ASYNC_LDS
            __builtin_amdgcn_global_load_async_to_lds_b128((gptr_b128)gp, (lptr_b128)lp, 0, 0);
#else
            *(v8bf*)lp = *(const v8bf*)gp;
#endif
        }
        // Stage B transposed: w2e[k][n] -> Bs[n][k], fp32 -> bf16.
        for (int idx = tid; idx < KCHUNK * (TILE_N / 4); idx += 256) {
            int k = idx >> 5, g = idx & 31;
            const float4 v = *(const float4*)(w2e + (size_t)(kc + k) * H_DIM + n0 + g * 4);
            Bs[g * 4 + 0][k] = f2bf(v.x); Bs[g * 4 + 1][k] = f2bf(v.y);
            Bs[g * 4 + 2][k] = f2bf(v.z); Bs[g * 4 + 3][k] = f2bf(v.w);
        }
#if HAVE_ASYNC_LDS
        __builtin_amdgcn_s_wait_asynccnt(0);
#endif
        __syncthreads();
#pragma unroll
        for (int ks = 0; ks < KCHUNK; ks += 32) {
            v16bf a0 = frag16(&As[mi0 * 16][0], lane, ks);
            v16bf a1 = frag16(&As[(mi0 + 1) * 16][0], lane, ks);
            v16bf b0 = frag16(&Bs[ni0 * 16][0], lane, ks);
            v16bf b1f = frag16(&Bs[(ni0 + 1) * 16][0], lane, ks);
            acc00 = WMMA_BF16(a0, b0, acc00);
            acc01 = WMMA_BF16(a0, b1f, acc01);
            acc10 = WMMA_BF16(a1, b0, acc10);
            acc11 = WMMA_BF16(a1, b1f, acc11);
        }
        __syncthreads();
    }
    int mrow0 = mi0 * 16 + (lane >> 4) * 8;
    int nc0 = n0 + ni0 * 16 + (lane & 15);
    v8f accs[2][2] = {{acc00, acc01}, {acc10, acc11}};
#pragma unroll
    for (int i = 0; i < 2; i++)
#pragma unroll
        for (int j = 0; j < 2; j++) {
            int nc = nc0 + j * 16;
            float bias = b2[e * H_DIM + nc];
#pragma unroll
            for (int v = 0; v < 8; v++) {
                int gr = rowBase + mrow0 + i * 16 + v;
                if (gr < rowEnd) {
                    int tok = rowTok[gr];
                    float wgt = rowW[gr];
                    atomicAdd(&out[(size_t)tok * H_DIM + nc],
                              wgt * (accs[i][j][v] + bias));
                }
            }
        }
}

extern "C" void kernel_launch(void* const* d_in, const int* in_sizes, int n_in,
                              void* d_out, int out_size, void* d_ws, size_t ws_size,
                              hipStream_t stream) {
    const float* hs = (const float*)d_in[0];
    const float* rw = (const float*)d_in[1];
    const float* rb = (const float*)d_in[2];
    const float* w1 = (const float*)d_in[3];
    const float* b1 = (const float*)d_in[4];
    const float* w2 = (const float*)d_in[5];
    const float* b2 = (const float*)d_in[6];
    float* out = (float*)d_out;

    // Workspace layout (needs ~64.2 MB; keeps hbuf 16B aligned).
    int* cnt    = (int*)d_ws;                 // [8]
    int* cnt2   = cnt + 8;                    // [8]
    int* eoff   = cnt2 + 8;                   // [8]
    int* rowTok = cnt + 24;                   // [8192]
    float* rowW = (float*)(rowTok + ROWS_TOTAL);
    int* tokExp = (int*)(rowW + ROWS_TOTAL);
    float* tokW = (float*)(tokExp + ROWS_TOTAL);
    bf16_t* hbuf = (bf16_t*)(tokW + ROWS_TOTAL); // [8192 * 4096] bf16

    router_k<<<TOKENS / 8, 256, 0, stream>>>(hs, rw, rb, tokExp, tokW);
    zero_counts_k<<<1, 64, 0, stream>>>(cnt, cnt2);
    count_k<<<ROWS_TOTAL / 256, 256, 0, stream>>>(tokExp, cnt);
    prefix_k<<<1, 32, 0, stream>>>(cnt, eoff);
    scatter_k<<<ROWS_TOTAL / 256, 256, 0, stream>>>(tokExp, tokW, eoff, cnt2, rowTok, rowW);
    zero_out_k<<<(TOKENS * H_DIM) / 256, 256, 0, stream>>>(out);

    dim3 g1(NE * 64, F_DIM / TILE_N);   // (512, 32); empty tiles early-exit
    ffn1_k<<<g1, 256, 0, stream>>>(hs, w1, b1, eoff, cnt, rowTok, hbuf);
    dim3 g2(NE * 64, H_DIM / TILE_N);   // (512, 8)
    ffn2_k<<<g2, 256, 0, stream>>>(hbuf, w2, b2, eoff, cnt, rowTok, rowW, out);
}
